// VectorQuantizerWithEntropy_3496103379398
// MI455X (gfx1250) — compile-verified, matrix operand order checked by reading, not run on hardware
//
#include <hip/hip_runtime.h>
#include <hip/hip_bf16.h>
#include <math.h>

typedef _Float16 half_t;
typedef __attribute__((ext_vector_type(16))) _Float16 v16h;
typedef __attribute__((ext_vector_type(8)))  _Float16 v8h;
typedef __attribute__((ext_vector_type(4)))  _Float16 v4h;
typedef __attribute__((ext_vector_type(8)))  float    v8f;
typedef __attribute__((ext_vector_type(4)))  float    v4f;
typedef __attribute__((ext_vector_type(4)))  unsigned v4u;
typedef __attribute__((ext_vector_type(8)))  int      v8i;
typedef __attribute__((ext_vector_type(4)))  int      v4i;

#define NUM_CODES 8192
#define CODE_DIM  512
#define N_VECS    32768              // 8*4096
#define ZQ_ELEMS  (N_VECS * CODE_DIM)

#define WAVES_PER_BLOCK 8
#define ROWS_PER_BLOCK  (WAVES_PER_BLOCK * 16)        // 128
#define TILE_BYTES      (16 * CODE_DIM * 2)           // 16 KB of f16 codebook per tile
#define ROW_STRIDE_LDS  (CODE_DIM * 2 + 16)           // 1040 B: 16 B pad per code row
#define TILE_LDS        (16 * ROW_STRIDE_LDS)         // 16640 B per staged tile

#define HAVE_TDM   __has_builtin(__builtin_amdgcn_tensor_load_to_lds)
#define HAVE_ASYNC __has_builtin(__builtin_amdgcn_global_load_async_to_lds_b128)

// ---------------------------------------------------------------------------
// Kernel 0: zero the histogram + loss accumulator (d_ws is poisoned 0xAA)
// ---------------------------------------------------------------------------
__global__ void vq_init(unsigned* __restrict__ counts, float* __restrict__ accum) {
    int i = blockIdx.x * 256 + threadIdx.x;
    if (i < NUM_CODES) counts[i] = 0u;
    if (i == 0) accum[0] = 0.0f;
}

// ---------------------------------------------------------------------------
// Kernel 1: embed f32 -> f16 (ws) and c_e = ||e||^2 / 2 per code
// ---------------------------------------------------------------------------
__global__ __launch_bounds__(128) void vq_prep_embed(
    const float* __restrict__ embed, half_t* __restrict__ ef16,
    float* __restrict__ ce)
{
    const int row = blockIdx.x;
    const int t   = threadIdx.x;
    const float* er = embed + (size_t)row * CODE_DIM;

    v4f e = *(const v4f*)(er + t * 4);
    v4h h;
    float s = 0.0f;
#pragma unroll
    for (int j = 0; j < 4; ++j) {
        h[j] = (half_t)e[j];
        s += e[j] * e[j];
    }
    *(v4h*)(ef16 + (size_t)row * CODE_DIM + t * 4) = h;

#pragma unroll
    for (int m = 16; m >= 1; m >>= 1) s += __shfl_down(s, m, 32);
    __shared__ float ls[4];
    if ((t & 31) == 0) ls[t >> 5] = s;
    __syncthreads();
    if (t == 0) ce[row] = 0.5f * (ls[0] + ls[1] + ls[2] + ls[3]);
}

// ---------------------------------------------------------------------------
// Tile staging: TDM (tensor_load_to_lds) primary, async-LDS loads secondary,
// plain load+ds_store fallback. One 16 KB contiguous tile (16 code rows) is
// staged with a 16 B pad after each 1 KB row (bank de-phasing for ds reads).
// ---------------------------------------------------------------------------
#if HAVE_TDM
__device__ __forceinline__ void tdm_issue(unsigned lds_off, const void* gsrc) {
    unsigned long long ga = (unsigned long long)(uintptr_t)gsrc;
    v4u g0;
    g0.x = 1u;                                             // count=1, user mode
    g0.y = lds_off;                                        // lds_addr
    g0.z = (unsigned)ga;                                   // global_addr[31:0]
    g0.w = (unsigned)((ga >> 32) & 0x01ffffffu) | (2u << 30); // addr[56:32] | type=2
    v8i g1;
    g1[0] = (3 << 16)            // data_size = 8 bytes
          | (1 << 20)            // pad_enable
          | (7 << 22)            // pad_interval: every 256 dwords (1 KB)
          | (3 << 25);           // pad_amount: 4 dwords (16 B)
    g1[1] = (int)(2048u << 16);  // tensor_dim0[15:0] = 2048 elems
    g1[2] = (int)(1u << 16);     // tensor_dim0 hi=0 | tensor_dim1[15:0]=1
    g1[3] = (int)(2048u << 16);  // tensor_dim1 hi=0 | tile_dim0=2048
    g1[4] = 1;                   // tile_dim1=1, tile_dim2=0
    g1[5] = 2048;                // tensor_dim0_stride lo
    g1[6] = 0;                   // stride hi | tensor_dim1_stride lo
    g1[7] = 0;
    v4i gz = {0, 0, 0, 0};
#if defined(__clang_major__) && (__clang_major__ >= 23)
    v8i gz8 = {0, 0, 0, 0, 0, 0, 0, 0};
    __builtin_amdgcn_tensor_load_to_lds(g0, g1, gz, gz, gz8, 0);
#else
    __builtin_amdgcn_tensor_load_to_lds(g0, g1, gz, gz, 0);
#endif
}
#endif

__device__ __forceinline__ void stage_tile(char* lds_base, unsigned lds_off,
                                           const half_t* gsrc, int tid, int wave) {
#if HAVE_TDM
    (void)lds_base; (void)tid;
    if (wave == 0) tdm_issue(lds_off, gsrc);
#elif HAVE_ASYNC
    (void)lds_off; (void)wave;
    const int row = tid >> 4, seg = tid & 15;              // 256 threads x 64 B
    const char* g = (const char*)gsrc + row * 1024 + seg * 64;
    char* l = lds_base + row * ROW_STRIDE_LDS + seg * 64;
#pragma unroll
    for (int j = 0; j < 4; ++j)
        __builtin_amdgcn_global_load_async_to_lds_b128(
            (const __attribute__((address_space(1))) void*)(g + 16 * j),
            (__attribute__((address_space(3))) void*)(l + 16 * j), 0, 0);
#else
    (void)lds_off; (void)wave;
    const int row = tid >> 4, seg = tid & 15;
    const v8h* g = (const v8h*)((const char*)gsrc + row * 1024 + seg * 64);
    v8h* l = (v8h*)(lds_base + row * ROW_STRIDE_LDS + seg * 64);
#pragma unroll
    for (int j = 0; j < 4; ++j) l[j] = g[j];
#endif
}

__device__ __forceinline__ void stage_wait(int wave) {
#if HAVE_TDM
    if (wave == 0) __builtin_amdgcn_s_wait_tensorcnt(0);
#elif HAVE_ASYNC
#if __has_builtin(__builtin_amdgcn_s_wait_asynccnt)
    __builtin_amdgcn_s_wait_asynccnt(0);
#endif
#endif
    __syncthreads();
}

// ---------------------------------------------------------------------------
// Kernel 2: fused WMMA GEMM + running argmax.
// score(row, code) = z.e - ||e||^2/2  (argmax == argmin of sq-distance).
// 256 threads = 8 waves; each wave owns 16 rows. The 8 waves share each
// 16-code tile staged in LDS by the TDM (double-buffered, TENSORcnt
// handshake). Inner loop uses one-chunk B lookahead (two live 8-VGPR B
// groups) so the next chunk's ds_load_b128 pair is in flight behind the
// current WMMA. __launch_bounds__(256,1) relaxes the occupancy target so
// the scheduler favors ILP over register-pressure serialization.
// ---------------------------------------------------------------------------
__global__ __launch_bounds__(256, 1) void vq_argmin(
    const float* __restrict__ z, const half_t* __restrict__ ef16,
    const float* __restrict__ ce, float* __restrict__ codes_out,
    unsigned* __restrict__ counts)
{
    const int tid  = threadIdx.x;
    const int lane = tid & 31;
    const int wave = tid >> 5;
    const int rowBase = (blockIdx.x * WAVES_PER_BLOCK + wave) * 16;
    const int col  = lane & 15;   // column within tile / A row within wave
    const int sel  = lane >> 4;   // which half of each K-chunk this lane holds
    const int row  = rowBase + col;

    __shared__ __attribute__((aligned(16))) char smem[2 * TILE_LDS];
    const unsigned lds0 = (unsigned)(unsigned long long)(uintptr_t)(void*)smem;

    // --- Register-resident A fragments (16 rows x 512 f16, ISA 16x32 striping)
    v16h a[16];
    const float* zr = z + (size_t)row * CODE_DIM;
#pragma unroll
    for (int kc = 0; kc < 16; ++kc) {
        const int k0 = kc * 32 + sel * 8;
        v4f f0 = *(const v4f*)(zr + k0);
        v4f f1 = *(const v4f*)(zr + k0 + 4);
        v4f f2 = *(const v4f*)(zr + k0 + 16);
        v4f f3 = *(const v4f*)(zr + k0 + 20);
        v16h av;
#pragma unroll
        for (int j = 0; j < 4; ++j) {
            av[j]      = (half_t)f0[j];
            av[j + 4]  = (half_t)f1[j];
            av[j + 8]  = (half_t)f2[j];
            av[j + 12] = (half_t)f3[j];
        }
        a[kc] = av;
    }

    float bval[8];
    int   bidx[8];
#pragma unroll
    for (int r = 0; r < 8; ++r) { bval[r] = -__builtin_inff(); bidx[r] = 0; }

    // Prologue: stage tile 0 into buffer 0
    stage_tile(smem, lds0, ef16, tid, wave);

    for (int tile = 0; tile < NUM_CODES / 16; ++tile) {
        stage_wait(wave);                       // tile data ready; prev buffer free

        const int nb = (tile + 1) & 1;
        if (tile + 1 < NUM_CODES / 16)
            stage_tile(smem + nb * TILE_LDS, lds0 + (unsigned)(nb * TILE_LDS),
                       ef16 + (size_t)(tile + 1) * 16 * CODE_DIM, tid, wave);

        const char* ebb = smem + (tile & 1) * TILE_LDS + col * ROW_STRIDE_LDS;
        v8f acc = {};

        // One-chunk B lookahead: two live B register groups.
        v8h c0 = *(const v8h*)(ebb + sel * 16);
        v8h c1 = *(const v8h*)(ebb + sel * 16 + 32);
#pragma unroll
        for (int kc = 0; kc < 16; ++kc) {
            v8h n0 = c0, n1 = c1;
            if (kc < 15) {
                const int k0 = (kc + 1) * 32 + sel * 8;
                n0 = *(const v8h*)(ebb + k0 * 2);
                n1 = *(const v8h*)(ebb + k0 * 2 + 32);
            }
            v16h bv;
#pragma unroll
            for (int e2 = 0; e2 < 8; ++e2) { bv[e2] = c0[e2]; bv[e2 + 8] = c1[e2]; }
            acc = __builtin_amdgcn_wmma_f32_16x16x32_f16(
                false, a[kc], false, bv, (short)0, acc, false, false);
            c0 = n0; c1 = n1;
        }

        const int cn = tile * 16 + col;
        const float cep = ce[cn];
#pragma unroll
        for (int r = 0; r < 8; ++r) {
            float v = acc[r] - cep;
            bool upd = v > bval[r];      // strict > keeps earliest code on ties
            bval[r] = upd ? v : bval[r];
            bidx[r] = upd ? cn : bidx[r];
        }
    }

    // Cross-lane argmax within each 16-lane half (C layout: slot r is row
    // rowBase+r on lanes 0-15 and rowBase+8+r on lanes 16-31).
#pragma unroll
    for (int r = 0; r < 8; ++r) {
        float v = bval[r];
        int   i = bidx[r];
#pragma unroll
        for (int m = 1; m <= 8; m <<= 1) {
            float ov = __shfl_xor(v, m, 32);
            int   oi = __shfl_xor(i, m, 32);
            if (ov > v || (ov == v && oi < i)) { v = ov; i = oi; }
        }
        if (col == 0) {
            const int orow = rowBase + sel * 8 + r;
            codes_out[orow] = (float)i;
            atomicAdd(&counts[i], 1u);
        }
    }
}

// ---------------------------------------------------------------------------
// Kernel 3: gather z_q = embed[code] (fp32) + commitment-loss partial sums.
// ---------------------------------------------------------------------------
__global__ __launch_bounds__(128) void vq_gather(
    const float* __restrict__ z, const float* __restrict__ embed,
    const float* __restrict__ codes_f, float* __restrict__ zq,
    float* __restrict__ commit_accum)
{
    const int row = blockIdx.x;
    const int t   = threadIdx.x;
    const int code = (int)codes_f[row];

    const float* er = embed + (size_t)code * CODE_DIM;
    const float* zr = z     + (size_t)row  * CODE_DIM;
    float*       qr = zq    + (size_t)row  * CODE_DIM;

    v4f e  = *(const v4f*)(er + t * 4);
    v4f zz = *(const v4f*)(zr + t * 4);
    *(v4f*)(qr + t * 4) = e;

    float s = 0.0f;
#pragma unroll
    for (int j = 0; j < 4; ++j) { float d = zz[j] - e[j]; s += d * d; }

#pragma unroll
    for (int m = 16; m >= 1; m >>= 1) s += __shfl_down(s, m, 32);
    __shared__ float ls[4];
    if ((t & 31) == 0) ls[t >> 5] = s;
    __syncthreads();
    if (t == 0) atomicAdd(commit_accum, ls[0] + ls[1] + ls[2] + ls[3]);
}

// ---------------------------------------------------------------------------
// Kernel 4: avg_probs, entropy, scalars. One 256-thread block.
// ---------------------------------------------------------------------------
__global__ __launch_bounds__(256) void vq_finalize(
    const unsigned* __restrict__ counts, const float* __restrict__ commit_accum,
    float* __restrict__ avg_probs, float* __restrict__ scalars)
{
    __shared__ float red[256];
    float s = 0.0f;
    for (int k = threadIdx.x; k < NUM_CODES; k += 256) {
        float p = (float)counts[k] * (1.0f / (float)N_VECS);
        avg_probs[k] = p;
        s += -p * logf(p + 1e-10f);
    }
    red[threadIdx.x] = s;
    __syncthreads();
    for (int off = 128; off > 0; off >>= 1) {
        if (threadIdx.x < off) red[threadIdx.x] += red[threadIdx.x + off];
        __syncthreads();
    }
    if (threadIdx.x == 0) {
        const float H = red[0];
        scalars[0] = commit_accum[0] / (float)((size_t)N_VECS * CODE_DIM); // commitment_loss
        scalars[1] = H;                                                    // code_entropy
        scalars[2] = H / logf((float)NUM_CODES);                           // normalized_entropy
    }
}

// ---------------------------------------------------------------------------
extern "C" void kernel_launch(void* const* d_in, const int* in_sizes, int n_in,
                              void* d_out, int out_size, void* d_ws, size_t ws_size,
                              hipStream_t stream) {
    const float* z     = (const float*)d_in[0];   // [8,4096,512]
    const float* embed = (const float*)d_in[1];   // [8192,512]

    // d_out layout (flat, return order): z_q | codes | 3 scalars | avg_probs
    float* out       = (float*)d_out;
    float* d_zq      = out;
    float* d_codes   = out + ZQ_ELEMS;
    float* d_scal    = out + ZQ_ELEMS + N_VECS;
    float* d_probs   = out + ZQ_ELEMS + N_VECS + 3;

    // ws layout: embed_f16 (8 MB) | c_e (32 KB) | counts (32 KB) | accum
    char* ws = (char*)d_ws;
    half_t*   ws_ef16  = (half_t*)ws;
    float*    ws_ce    = (float*)(ws + (size_t)NUM_CODES * CODE_DIM * sizeof(half_t));
    unsigned* ws_cnt   = (unsigned*)((char*)ws_ce + NUM_CODES * sizeof(float));
    float*    ws_accum = (float*)((char*)ws_cnt + NUM_CODES * sizeof(unsigned));

    vq_init<<<(NUM_CODES + 255) / 256, 256, 0, stream>>>(ws_cnt, ws_accum);
    vq_prep_embed<<<NUM_CODES, 128, 0, stream>>>(embed, ws_ef16, ws_ce);
    vq_argmin<<<N_VECS / ROWS_PER_BLOCK, 256, 0, stream>>>(z, ws_ef16, ws_ce, d_codes, ws_cnt);
    vq_gather<<<N_VECS, 128, 0, stream>>>(z, embed, d_codes, d_zq, ws_accum);
    vq_finalize<<<1, 256, 0, stream>>>(ws_cnt, ws_accum, d_probs, d_scal);
}
